// decoder_layer_67207648247921
// MI455X (gfx1250) — matrix operand order repaired and checked by reading, not compiled, not run on hardware
//
#include <hip/hip_runtime.h>
#include <hip/hip_bf16.h>
#include <math.h>

typedef __attribute__((ext_vector_type(16))) __bf16 bf16x16;
typedef __attribute__((ext_vector_type(8)))  float  v8f;

#define DEVINL __device__ __forceinline__

union ABFrag { bf16x16 v; float4 q[2]; };

// A-fragment (16x32, 16-bit): lane row = lane&15; elems 0..7 = K[g*8..g*8+7],
// elems 8..15 = K[16+g*8 .. 16+g*8+7], g = lane>>4.  Two b128 loads.
DEVINL bf16x16 load_a_frag(const __bf16* base, int ld, int lane) {
  const __bf16* p = base + (size_t)(lane & 15) * ld + ((lane >> 4) << 3);
  ABFrag u;
  u.q[0] = *reinterpret_cast<const float4*>(p);
  u.q[1] = *reinterpret_cast<const float4*>(p + 16);
  return u.v;
}

// B-fragment (32x16, 16-bit) from transposed storage Bt[n][k]:
// lane col n = lane&15; elems 0..7 = K[g*8..g*8+7], elems 8..15 = K[16+g*8 .. 16+g*8+7].
DEVINL bf16x16 load_b_frag(const __bf16* base, int ld, int lane) {
  const __bf16* p = base + (size_t)(lane & 15) * ld + ((lane >> 4) << 3);
  ABFrag u;
  u.q[0] = *reinterpret_cast<const float4*>(p);
  u.q[1] = *reinterpret_cast<const float4*>(p + 16);
  return u.v;
}

DEVINL v8f wmma_bf16(bf16x16 a, bf16x16 b, v8f c) {
  return __builtin_amdgcn_wmma_f32_16x16x32_bf16(false, a, false, b, (short)0, c,
                                                 false, false);
}

// CDNA5 async copy global -> LDS (ASYNCcnt-tracked), 16B per lane.
// Low 32 bits of a generic LDS pointer == byte offset within the WG's LDS.
DEVINL void async_b128(void* lds, const void* g) {
  asm volatile("global_load_async_to_lds_b128 %0, %1, off"
               :: "v"((unsigned)(size_t)lds),
                  "v"((unsigned long long)(size_t)g)
               : "memory");
}
DEVINL void wait_async0() { asm volatile("s_wait_asynccnt 0" ::: "memory"); }

// row reductions across the 16-lane half (C-layout: row fixed within group)
DEVINL float red_max16(float v) {
#pragma unroll
  for (int m = 1; m <= 8; m <<= 1) v = fmaxf(v, __shfl_xor(v, m, 32));
  return v;
}
DEVINL float red_sum16(float v) {
#pragma unroll
  for (int m = 1; m <= 8; m <<= 1) v += __shfl_xor(v, m, 32);
  return v;
}

// ---------------------------------------------------------------- converts
__global__ __launch_bounds__(256) void cvt_f32_bf16_kernel(const float* __restrict__ in,
                                                           __bf16* __restrict__ out,
                                                           int n) {
  int i = blockIdx.x * blockDim.x + threadIdx.x;
  if (i < n) out[i] = (__bf16)in[i];
}

// W[K][N] fp32 -> Wt[N][K] bf16
__global__ __launch_bounds__(256) void transpose_cvt_kernel(const float* __restrict__ W,
                                                            __bf16* __restrict__ Wt,
                                                            int K, int N) {
  int i = blockIdx.x * blockDim.x + threadIdx.x;
  if (i >= K * N) return;
  int k = i / N, n = i - k * N;
  Wt[(size_t)n * K + k] = (__bf16)W[i];
}

// ---------------------------------------------------------------- WMMA GEMM
// Block = 8 waves, block tile 128M x 128N, wave tile 16M x 128N.
// B tile (128n x 32k, 8KB) staged in LDS via async copies, double-buffered,
// shared by all 8 waves.
// MODE 0: QKV  -> Q,K packed bf16 [M][2048], V scattered to Vt[b][h][d][t]
// MODE 1: RES  -> outf = acc + bias + res   (fp32)
// MODE 2: RELU -> outb = relu(acc + bias)   (bf16)
template <int MODE>
__global__ __launch_bounds__(256) void wmma_gemm_kernel(
    const __bf16* __restrict__ A, const __bf16* __restrict__ Bt,
    const float* __restrict__ bias, const float* __restrict__ res,
    float* __restrict__ outf, __bf16* __restrict__ outb, __bf16* __restrict__ vt,
    int M, int N, int K) {
  __shared__ __bf16 bsm[2][128 * 32];
  int tid = threadIdx.x;
  int lane = tid & 31, wslot = tid >> 5;
  int tiles_n = N >> 7;
  int bm = (blockIdx.x / tiles_n) << 7;
  int bn = (blockIdx.x % tiles_n) << 7;
  int tm = bm + (wslot << 4);

  v8f zero = {0.f, 0.f, 0.f, 0.f, 0.f, 0.f, 0.f, 0.f};
  v8f acc[8] = {zero, zero, zero, zero, zero, zero, zero, zero};

  // prologue stage of buffer 0 (each thread copies 2 x 16B)
#pragma unroll
  for (int c = 0; c < 2; ++c) {
    int chunk = tid + c * 256;                 // 0..511
    int n = chunk >> 2, sub = (chunk & 3) << 3;
    async_b128(&bsm[0][n * 32 + sub], Bt + (size_t)(bn + n) * K + sub);
  }

  int nsteps = K >> 5;
  for (int i = 0; i < nsteps; ++i) {
    int cur = i & 1;
    wait_async0();
    __syncthreads();
    if (i + 1 < nsteps) {
      int k0n = (i + 1) << 5;
#pragma unroll
      for (int c = 0; c < 2; ++c) {
        int chunk = tid + c * 256;
        int n = chunk >> 2, sub = (chunk & 3) << 3;
        async_b128(&bsm[cur ^ 1][n * 32 + sub],
                   Bt + (size_t)(bn + n) * K + k0n + sub);
      }
    }
    bf16x16 a = load_a_frag(A + (size_t)tm * K + (i << 5), K, lane);
#pragma unroll
    for (int j = 0; j < 8; ++j) {
      bf16x16 b = load_b_frag(&bsm[cur][(j << 4) * 32], 32, lane);
      acc[j] = wmma_bf16(a, b, acc[j]);
    }
  }

  int gm = (lane >> 4) << 3, ln = lane & 15;
#pragma unroll
  for (int j = 0; j < 8; ++j) {
#pragma unroll
    for (int r = 0; r < 8; ++r) {
      int m = tm + gm + r;
      int n = bn + j * 16 + ln;
      float v = acc[j][r] + bias[n];
      if (MODE == 1) {
        v += res[(size_t)m * N + n];
        outf[(size_t)m * N + n] = v;
      } else if (MODE == 2) {
        outb[(size_t)m * N + n] = (__bf16)fmaxf(v, 0.f);
      } else {  // QKV
        if (n < 2048) {
          outb[(size_t)m * 2048 + n] = (__bf16)v;  // Q cols 0..1023, K cols 1024..2047
        } else {
          int b_ = m >> 11, t = m & 2047;
          int nn = n - 2048, h = nn >> 6, d = nn & 63;
          vt[((((size_t)b_ * 16 + h) << 6) + d) * 2048 + t] = (__bf16)v;
        }
      }
    }
  }
}

// ---------------------------------------------------------------- attention
// Block = 8 waves covering 128 consecutive queries of one (b,h); K/V tiles
// staged in LDS (async, double-buffered) and shared by all waves.
__global__ __launch_bounds__(256) void attention_kernel(
    const __bf16* __restrict__ qk,   // [B*T][2048]: Q | K
    const __bf16* __restrict__ vt,   // [B][H][64][2048]
    __bf16* __restrict__ attn_out) { // [B*T][1024] bf16
  constexpr int T = 2048, H = 16, LD = 2048;
  __shared__ __bf16 ksm[2][32 * 64];  // [kt][d]
  __shared__ __bf16 vsm[2][64 * 32];  // [d][kt]
  __shared__ __bf16 psm[8][16 * 32];  // per-wave P tile

  int tid = threadIdx.x;
  int lane = tid & 31, wslot = tid >> 5;
  int blk = blockIdx.x;
  int qblk = blk & (T / 128 - 1);
  int bh = blk / (T / 128);
  int h = bh & (H - 1);
  int b = bh >> 4;
  int bq0 = qblk << 7;
  int q0 = bq0 + (wslot << 4);

  const __bf16* Qb = qk + (size_t)b * T * LD + h * 64;
  const __bf16* Kb = Qb + 1024;
  const __bf16* Vb = vt + (size_t)(b * H + h) * 64 * T;

  bf16x16 qa0 = load_a_frag(Qb + (size_t)q0 * LD, LD, lane);
  bf16x16 qa1 = load_a_frag(Qb + (size_t)q0 * LD + 32, LD, lane);

  v8f zero = {0.f, 0.f, 0.f, 0.f, 0.f, 0.f, 0.f, 0.f};
  v8f o[4] = {zero, zero, zero, zero};
  float mrow[8], lrow[8];
#pragma unroll
  for (int r = 0; r < 8; ++r) { mrow[r] = -__builtin_inff(); lrow[r] = 0.f; }

  __bf16* P = psm[wslot];
  const float sc = 0.125f;  // 1/sqrt(64)

  // cooperative K/V tile stage for key block starting at kt0
  auto stage = [&](int buf, int kt0) {
    {  // K tile: 32 rows x 64 halves
      int row = tid >> 3, sub = (tid & 7) << 3;
      async_b128(&ksm[buf][row * 64 + sub], Kb + (size_t)(kt0 + row) * LD + sub);
    }
    {  // V tile: 64 rows (d) x 32 halves (kt)
      int d = tid >> 2, sub = (tid & 3) << 3;
      async_b128(&vsm[buf][d * 32 + sub], Vb + (size_t)d * T + kt0 + sub);
    }
  };

  int nsteps = (bq0 + 128) >> 5;  // uniform across block
  stage(0, 0);
  for (int i = 0; i < nsteps; ++i) {
    int kt0 = i << 5;
    int cur = i & 1;
    wait_async0();
    __syncthreads();
    if (i + 1 < nsteps) stage(cur ^ 1, kt0 + 32);

    if (kt0 < q0 + 16) {  // wave-uniform: EXEC stays all-ones
      v8f s0 = zero, s1 = zero;
      s0 = wmma_bf16(qa0, load_b_frag(&ksm[cur][0], 64, lane), s0);
      s0 = wmma_bf16(qa1, load_b_frag(&ksm[cur][32], 64, lane), s0);
      s1 = wmma_bf16(qa0, load_b_frag(&ksm[cur][16 * 64], 64, lane), s1);
      s1 = wmma_bf16(qa1, load_b_frag(&ksm[cur][16 * 64 + 32], 64, lane), s1);

      bool need_mask = (kt0 + 31 > q0);
      int mrel_base = (lane >> 4) << 3;
      int ln = lane & 15;
#pragma unroll
      for (int r = 0; r < 8; ++r) {
        float a = s0[r] * sc, c = s1[r] * sc;
        if (need_mask) {
          int q = q0 + mrel_base + r;
          if (kt0 + ln > q) a = -__builtin_inff();
          if (kt0 + 16 + ln > q) c = -__builtin_inff();
        }
        float mx = red_max16(fmaxf(a, c));
        float mnew = fmaxf(mrow[r], mx);
        float alpha = __expf(mrow[r] - mnew);
        float p0 = __expf(a - mnew);
        float p1 = __expf(c - mnew);
        mrow[r] = mnew;
        lrow[r] = lrow[r] * alpha + red_sum16(p0 + p1);
        o[0][r] *= alpha; o[1][r] *= alpha; o[2][r] *= alpha; o[3][r] *= alpha;
        int mrel = mrel_base + r;
        P[mrel * 32 + ln] = (__bf16)p0;
        P[mrel * 32 + 16 + ln] = (__bf16)p1;
      }
      __builtin_amdgcn_wave_barrier();  // keep P stores before P loads

      bf16x16 pa = load_a_frag(P, 32, lane);
#pragma unroll
      for (int j = 0; j < 4; ++j) {
        bf16x16 vb = load_b_frag(&vsm[cur][(j << 4) * 32], 32, lane);
        o[j] = wmma_bf16(pa, vb, o[j]);
      }
    }
  }

  int mrel_base = (lane >> 4) << 3;
  int ln = lane & 15;
#pragma unroll
  for (int j = 0; j < 4; ++j) {
#pragma unroll
    for (int r = 0; r < 8; ++r) {
      float v = o[j][r] / lrow[r];
      int m = q0 + mrel_base + r;
      int d = j * 16 + ln;
      attn_out[((size_t)(b * T + m)) * 1024 + h * 64 + d] = (__bf16)v;
    }
  }
}

// ---------------------------------------------------------------- layernorm
__global__ __launch_bounds__(256) void layernorm_kernel(
    const float* __restrict__ in, const float* __restrict__ g,
    const float* __restrict__ be, float* __restrict__ out_f,
    __bf16* __restrict__ out_b, int E) {
  int row = blockIdx.x, tid = threadIdx.x;
  const float* p = in + (size_t)row * E;
  float s = 0.f, ss = 0.f;
  for (int i = tid; i < E; i += blockDim.x) {
    float v = p[i];
    s += v;
    ss += v * v;
  }
  __shared__ float r1[256], r2[256];
  r1[tid] = s; r2[tid] = ss;
  __syncthreads();
  for (int o = 128; o > 0; o >>= 1) {
    if (tid < o) { r1[tid] += r1[tid + o]; r2[tid] += r2[tid + o]; }
    __syncthreads();
  }
  float mean = r1[0] / E;
  float var = r2[0] / E - mean * mean;
  float rstd = rsqrtf(var + 1e-5f);
  for (int i = tid; i < E; i += blockDim.x) {
    float v = (p[i] - mean) * rstd * g[i] + be[i];
    if (out_f) out_f[(size_t)row * E + i] = v;
    if (out_b) out_b[(size_t)row * E + i] = (__bf16)v;
  }
}

// ---------------------------------------------------------------- launcher
extern "C" void kernel_launch(void* const* d_in, const int* in_sizes, int n_in,
                              void* d_out, int out_size, void* d_ws, size_t ws_size,
                              hipStream_t stream) {
  (void)in_sizes; (void)n_in; (void)out_size; (void)ws_size;
  const float* x    = (const float*)d_in[0];
  const float* Wqkv = (const float*)d_in[1];
  const float* bqkv = (const float*)d_in[2];
  const float* Wout = (const float*)d_in[3];
  const float* bout = (const float*)d_in[4];
  const float* g1   = (const float*)d_in[5];
  const float* be1  = (const float*)d_in[6];
  const float* Wff1 = (const float*)d_in[7];
  const float* bff1 = (const float*)d_in[8];
  const float* Wff2 = (const float*)d_in[9];
  const float* bff2 = (const float*)d_in[10];
  const float* g2   = (const float*)d_in[11];
  const float* be2  = (const float*)d_in[12];

  constexpr int Bb = 2, T = 2048, E = 1024, H = 16, D = 64, F = 2048, E3 = 3072;
  constexpr int M = Bb * T;  // 4096

  char* ws = (char*)d_ws;
  size_t off = 0;
  auto take = [&](size_t bytes) -> char* {
    char* p = ws + off;
    off += (bytes + 255) & ~(size_t)255;
    return p;
  };

  __bf16* xb    = (__bf16*)take((size_t)M * E * 2);
  __bf16* Wqkvt = (__bf16*)take((size_t)E3 * E * 2);
  __bf16* Woutt = (__bf16*)take((size_t)E * E * 2);
  __bf16* Wff1t = (__bf16*)take((size_t)F * E * 2);
  __bf16* Wff2t = (__bf16*)take((size_t)E * F * 2);
  __bf16* qkb   = (__bf16*)take((size_t)M * 2048 * 2);        // Q|K packed
  __bf16* vtb   = (__bf16*)take((size_t)Bb * H * D * T * 2);  // V transposed
  __bf16* attnb = (__bf16*)take((size_t)M * E * 2);
  float*  y1    = (float*)take((size_t)M * E * 4);
  float*  x1f   = (float*)take((size_t)M * E * 4);
  __bf16* x1b   = (__bf16*)take((size_t)M * E * 2);
  __bf16* ffb   = (__bf16*)take((size_t)M * F * 2);
  float*  y2    = y1;  // safe reuse: LN1 consumed y1 before FF2 writes

  cvt_f32_bf16_kernel<<<(M * E) / 256, 256, 0, stream>>>(x, xb, M * E);
  transpose_cvt_kernel<<<(E * E3 + 255) / 256, 256, 0, stream>>>(Wqkv, Wqkvt, E, E3);
  transpose_cvt_kernel<<<(E * E + 255) / 256, 256, 0, stream>>>(Wout, Woutt, E, E);
  transpose_cvt_kernel<<<(E * F + 255) / 256, 256, 0, stream>>>(Wff1, Wff1t, E, F);
  transpose_cvt_kernel<<<(F * E + 255) / 256, 256, 0, stream>>>(Wff2, Wff2t, F, E);

  // QKV projection
  wmma_gemm_kernel<0><<<(M / 128) * (E3 / 128), 256, 0, stream>>>(
      xb, Wqkvt, bqkv, nullptr, nullptr, qkb, vtb, M, E3, E);

  // flash attention
  attention_kernel<<<Bb * H * (T / 128), 256, 0, stream>>>(qkb, vtb, attnb);

  // out projection + residual (fp32)
  wmma_gemm_kernel<1><<<(M / 128) * (E / 128), 256, 0, stream>>>(
      attnb, Woutt, bout, x, y1, nullptr, nullptr, M, E, E);

  // LN1 -> fp32 (residual path) + bf16 (FF input)
  layernorm_kernel<<<M, 256, 0, stream>>>(y1, g1, be1, x1f, x1b, E);

  // FF1 + ReLU (bf16 out)
  wmma_gemm_kernel<2><<<(M / 128) * (F / 128), 256, 0, stream>>>(
      x1b, Wff1t, bff1, nullptr, nullptr, ffb, nullptr, M, F, E);

  // FF2 + residual (fp32)
  wmma_gemm_kernel<1><<<(M / 128) * (E / 128), 256, 0, stream>>>(
      ffb, Wff2t, bff2, x1f, y2, nullptr, nullptr, M, E, F);

  // LN2 -> final output
  layernorm_kernel<<<M, 256, 0, stream>>>(y2, g2, be2, (float*)d_out, nullptr, E);
}
